// FullAttnModule_80779744903217
// MI455X (gfx1250) — matrix operand order repaired and checked by reading, not compiled, not run on hardware
//
#include <hip/hip_runtime.h>
#include <hip/hip_bf16.h>

// ---------------------------------------------------------------------------
// Types matching CDNA5 WMMA builtin signatures
// ---------------------------------------------------------------------------
typedef __attribute__((ext_vector_type(16))) __bf16 v16bf;
typedef __attribute__((ext_vector_type(8)))  __bf16 v8bf;
typedef __attribute__((ext_vector_type(4)))  __bf16 v4bf;
typedef __attribute__((ext_vector_type(2)))  __bf16 v2bf;
typedef __attribute__((ext_vector_type(8)))  float  v8f;
typedef __attribute__((ext_vector_type(4)))  float  v4f;
typedef __attribute__((ext_vector_type(4)))  unsigned int v4u;

#define DIM      1024
#define HEADS    16
#define HEAD_DIM 64
#define NQ       2048
#define NK       2048
#define ATT_SCALE 0.125f   // 64^-0.5

__device__ __forceinline__ int lane_id() { return threadIdx.x & 31; }

// ---------------------------------------------------------------------------
// Row-of-16 cross-lane reductions via v_permlane16_b32 (pure VALU; matches
// the 16-lane halves of the WMMA C layout; avoids ds_bpermute on the DS pipe)
// ---------------------------------------------------------------------------
__device__ __forceinline__ float permlane16_f(float x, unsigned s0, unsigned s1) {
  int xi = __builtin_bit_cast(int, x);
  int r  = __builtin_amdgcn_permlane16(xi, xi, (int)s0, (int)s1, false, false);
  return __builtin_bit_cast(float, r);
}
// selector nibble j of s0 = target lane for lane j (row-relative), s1 for lanes 8-15
__device__ __forceinline__ float rowmax16(float x) {
  x = fmaxf(x, permlane16_f(x, 0x67452301u, 0xEFCDAB89u));  // xor 1
  x = fmaxf(x, permlane16_f(x, 0x54761032u, 0xDCFE98BAu));  // xor 2
  x = fmaxf(x, permlane16_f(x, 0x32107654u, 0xBA98FEDCu));  // xor 4
  x = fmaxf(x, permlane16_f(x, 0xFEDCBA98u, 0x76543210u));  // xor 8
  return x;
}
__device__ __forceinline__ float rowsum16(float x) {
  x += permlane16_f(x, 0x67452301u, 0xEFCDAB89u);
  x += permlane16_f(x, 0x54761032u, 0xDCFE98BAu);
  x += permlane16_f(x, 0x32107654u, 0xBA98FEDCu);
  x += permlane16_f(x, 0xFEDCBA98u, 0x76543210u);
  return x;
}

// ---------------------------------------------------------------------------
// Fragment loaders (wave32, 16x16x32 bf16 WMMA)
// ---------------------------------------------------------------------------
__device__ __forceinline__ v16bf frag_a(const __bf16* base, int row, int stride, int koff) {
  const int l = lane_id();
  const __bf16* p = base + (size_t)(row + (l & 15)) * stride + koff + ((l & 16) ? 8 : 0);
  v8bf lo = *(const v8bf*)(p);
  v8bf hi = *(const v8bf*)(p + 16);
  v16bf f;
#pragma unroll
  for (int i = 0; i < 8; ++i) { f[i] = lo[i]; f[i + 8] = hi[i]; }
  return f;
}

__device__ __forceinline__ v16bf frag_b(const __bf16* baseT, int col, int stride, int koff) {
  const int l = lane_id();
  const __bf16* p = baseT + (size_t)(col + (l & 15)) * stride + koff + ((l & 16) ? 8 : 0);
  v8bf lo = *(const v8bf*)(p);
  v8bf hi = *(const v8bf*)(p + 16);
  v16bf f;
#pragma unroll
  for (int i = 0; i < 8; ++i) { f[i] = lo[i]; f[i + 8] = hi[i]; }
  return f;
}

#define WMMA_BF16(A, B, C) \
  __builtin_amdgcn_wmma_f32_16x16x32_bf16(false, (A), false, (B), (short)0, (C), false, false)

// gfx1250 async copy: global -> LDS, 16 bytes per lane, tracked by ASYNCcnt.
__device__ __forceinline__ void async_copy_b128(unsigned lds_off, const void* gptr) {
  asm volatile("global_load_async_to_lds_b128 %0, %1, off"
               :: "v"(lds_off), "v"(gptr) : "memory");
}
__device__ __forceinline__ void wait_async0() {
  asm volatile("s_wait_asynccnt 0" ::: "memory");
}

// ---------------------------------------------------------------------------
// GEMM + bias:  Out[M,N] = A[M,K] * W[K,N] + bias[N]
// Block tile 128x64, 256 threads = 8 waves (4x2), wave tile 32x32.
// ---------------------------------------------------------------------------
template <bool A_BF16, bool OUT_BF16>
__global__ __launch_bounds__(256) void gemm_bias_kernel(
    const void* __restrict__ Aptr, const float* __restrict__ W,
    const float* __restrict__ bias, void* __restrict__ Outptr,
    int M, int N, int K) {
  __shared__ __attribute__((aligned(16))) __bf16 Asm[128 * 40];
  __shared__ __attribute__((aligned(16))) __bf16 BsmT[64 * 40];

  const int tid = threadIdx.x;
  const int m0 = blockIdx.x * 128;
  const int n0 = blockIdx.y * 64;
  const int w  = tid >> 5;
  const int wm = (w & 3) * 32;
  const int wn = (w >> 2) * 32;
  const int l  = lane_id();
  const int half = l >> 4;

  v8f acc00 = {}, acc01 = {}, acc10 = {}, acc11 = {};

  for (int kb = 0; kb < K; kb += 32) {
    // ---- stage A tile 128x32 as two 16B chunks per thread ----
    {
      const int r = tid >> 1, c = (tid & 1) * 16;
      __bf16* dst = &Asm[r * 40 + c];
      if (A_BF16) {
        const v4u* src = (const v4u*)((const __bf16*)Aptr + (size_t)(m0 + r) * K + kb + c);
        ((v4u*)dst)[0] = src[0];
        ((v4u*)dst)[1] = src[1];
      } else {
        const float* src = (const float*)Aptr + (size_t)(m0 + r) * K + kb + c;
        v8bf t0, t1;
#pragma unroll
        for (int i = 0; i < 8; ++i) { t0[i] = (__bf16)src[i]; t1[i] = (__bf16)src[i + 8]; }
        ((v8bf*)dst)[0] = t0;
        ((v8bf*)dst)[1] = t1;
      }
    }
    // ---- stage W tile 32x64 transposed, packed pairs along k ----
    {
      const int k = (tid & 15) * 2;   // pair of k rows
      const int c = (tid >> 4) * 4;   // 4 columns
      const float* s0 = W + (size_t)(kb + k) * N + n0 + c;
      v4f r0 = *(const v4f*)s0;
      v4f r1 = *(const v4f*)(s0 + N);
      if (kb + 32 < K) __builtin_prefetch(s0 + 32 * N, 0, 1);
#pragma unroll
      for (int i = 0; i < 4; ++i) {
        v2bf pk; pk[0] = (__bf16)r0[i]; pk[1] = (__bf16)r1[i];
        *(v2bf*)(&BsmT[(c + i) * 40 + k]) = pk;
      }
    }
    __syncthreads();

    v16bf a0 = frag_a(Asm, wm, 40, 0);
    v16bf a1 = frag_a(Asm, wm + 16, 40, 0);
    v16bf b0 = frag_b(BsmT, wn, 40, 0);
    v16bf b1 = frag_b(BsmT, wn + 16, 40, 0);
    acc00 = WMMA_BF16(a0, b0, acc00);
    acc01 = WMMA_BF16(a0, b1, acc01);
    acc10 = WMMA_BF16(a1, b0, acc10);
    acc11 = WMMA_BF16(a1, b1, acc11);
    __syncthreads();
  }

  // ---- epilogue: C layout lane l, vgpr j -> row j+8*half, col l&15 ----
#pragma unroll
  for (int ti = 0; ti < 2; ++ti) {
#pragma unroll
    for (int tj = 0; tj < 2; ++tj) {
      const v8f& a = ti == 0 ? (tj == 0 ? acc00 : acc01) : (tj == 0 ? acc10 : acc11);
#pragma unroll
      for (int j = 0; j < 8; ++j) {
        const int row = m0 + wm + ti * 16 + j + half * 8;
        const int col = n0 + wn + tj * 16 + (l & 15);
        const float v = a[j] + bias[col];
        if (OUT_BF16) ((__bf16*)Outptr)[(size_t)row * N + col] = (__bf16)v;
        else          ((float*)Outptr)[(size_t)row * N + col]  = v;
      }
    }
  }
}

// ---------------------------------------------------------------------------
// Flash attention: one workgroup = 128 q-rows of one (batch, head).
// 8 waves x 16 q-rows. K tile via async global->LDS; V tile transposed with
// permuted contraction index pi(k) = (k&15)*4 + (k>>4) -> packed b64 stores.
// Online softmax: row max via permlane16 butterflies; row sum kept as
// per-lane partials (alpha is uniform per 16-lane half) and reduced once.
// ---------------------------------------------------------------------------
__global__ __launch_bounds__(256) void attn_kernel(
    const __bf16* __restrict__ Q, const __bf16* __restrict__ Kd,
    const __bf16* __restrict__ Vd, const float* __restrict__ bias,
    __bf16* __restrict__ O) {
  __shared__ __attribute__((aligned(16))) __bf16 Kt[64 * 72];      // [k][c]
  __shared__ __attribute__((aligned(16))) __bf16 Vt[64 * 72];      // [c][pi(k)]
  __shared__ __attribute__((aligned(16))) __bf16 Ps[8 * 16 * 72];  // per-wave P scratch

  const int tid  = threadIdx.x;
  const int w    = tid >> 5;
  const int l    = lane_id();
  const int half = l >> 4;
  const int b    = blockIdx.z;
  const int h    = blockIdx.y;
  const int q0   = blockIdx.x * 128 + w * 16;

  const __bf16* Qb = Q  + (size_t)b * NQ * DIM + h * HEAD_DIM;
  const __bf16* Kb = Kd + (size_t)b * NK * DIM + h * HEAD_DIM;
  const __bf16* Vb = Vd + (size_t)b * NK * DIM + h * HEAD_DIM;
  const float* biasb = bias + (size_t)b * NQ * NK;

  v16bf qf0 = frag_a(Qb, q0, DIM, 0);
  v16bf qf1 = frag_a(Qb, q0, DIM, 32);

  v8f o0 = {}, o1 = {}, o2 = {}, o3 = {};
  float mrow[8], lpart[8];     // lpart: per-lane partial row sums
#pragma unroll
  for (int j = 0; j < 8; ++j) { mrow[j] = -3.0e38f; lpart[j] = 0.f; }

  __bf16* Pw = &Ps[w * 16 * 72];

  for (int kb = 0; kb < NK; kb += 64) {
    // ---- K tile: async bf16 copy global->LDS (ASYNCcnt) ----
    {
      const int k = tid >> 2, c = (tid & 3) * 16;
      const __bf16* ks = Kb + (size_t)(kb + k) * DIM + c;
      const unsigned lo = (unsigned)(size_t)(&Kt[k * 72 + c]);
      async_copy_b128(lo, ks);
      async_copy_b128(lo + 16, ks + 8);
    }
    // ---- V^T tile with permuted k ----
    {
      const int n  = tid & 15;
      const int cb = (tid >> 4) * 4;
      v4bf rows[4];
#pragma unroll
      for (int i = 0; i < 4; ++i)
        rows[i] = *(const v4bf*)(Vb + (size_t)(kb + n + i * 16) * DIM + cb);
#pragma unroll
      for (int i = 0; i < 4; ++i) {
        v4bf pk;
        pk[0] = rows[0][i]; pk[1] = rows[1][i]; pk[2] = rows[2][i]; pk[3] = rows[3][i];
        *(v4bf*)(&Vt[(cb + i) * 72 + n * 4]) = pk;
      }
    }
    wait_async0();
    __syncthreads();

    // ---- S = Q K^T ----
    v8f s[4];
#pragma unroll
    for (int kt = 0; kt < 4; ++kt) {
      v16bf bk0 = frag_b(Kt, kt * 16, 72, 0);
      v16bf bk1 = frag_b(Kt, kt * 16, 72, 32);
      v8f z = {};
      z = WMMA_BF16(qf0, bk0, z);
      z = WMMA_BF16(qf1, bk1, z);
      s[kt] = z;
    }

    // ---- scale + bias ----
#pragma unroll
    for (int kt = 0; kt < 4; ++kt) {
      const int kc = kb + kt * 16 + (l & 15);
#pragma unroll
      for (int j = 0; j < 8; ++j) {
        const int qr = q0 + j + half * 8;
        s[kt][j] = s[kt][j] * ATT_SCALE + biasb[(size_t)qr * NK + kc];
      }
    }

    // ---- online softmax: max reduce (VALU permlane), per-lane partial sums ----
#pragma unroll
    for (int j = 0; j < 8; ++j) {
      float mx = rowmax16(fmaxf(fmaxf(s[0][j], s[1][j]), fmaxf(s[2][j], s[3][j])));
      const float mnew  = fmaxf(mrow[j], mx);
      const float alpha = __expf(mrow[j] - mnew);
      mrow[j] = mnew;
      float rs = 0.f;
#pragma unroll
      for (int kt = 0; kt < 4; ++kt) {
        const float p = __expf(s[kt][j] - mnew);
        s[kt][j] = p;
        rs += p;
      }
      lpart[j] = lpart[j] * alpha + rs;   // per-lane partial; reduced once at end
      o0[j] *= alpha; o1[j] *= alpha; o2[j] *= alpha; o3[j] *= alpha;
    }

    // ---- P -> per-wave scratch (packed b64 per row via permuted k) ----
#pragma unroll
    for (int j = 0; j < 8; ++j) {
      v4bf pk;
      pk[0] = (__bf16)s[0][j]; pk[1] = (__bf16)s[1][j];
      pk[2] = (__bf16)s[2][j]; pk[3] = (__bf16)s[3][j];
      *(v4bf*)(&Pw[(j + half * 8) * 72 + (l & 15) * 4]) = pk;
    }
    asm volatile("s_wait_dscnt 0" ::: "memory");

    // ---- O += P V ----
    v16bf pa0 = frag_a(Pw, 0, 72, 0);
    v16bf pa1 = frag_a(Pw, 0, 72, 32);
    {
      v16bf bv0 = frag_b(Vt, 0, 72, 0),  bv1 = frag_b(Vt, 0, 72, 32);
      o0 = WMMA_BF16(pa0, bv0, o0); o0 = WMMA_BF16(pa1, bv1, o0);
    }
    {
      v16bf bv0 = frag_b(Vt, 16, 72, 0), bv1 = frag_b(Vt, 16, 72, 32);
      o1 = WMMA_BF16(pa0, bv0, o1); o1 = WMMA_BF16(pa1, bv1, o1);
    }
    {
      v16bf bv0 = frag_b(Vt, 32, 72, 0), bv1 = frag_b(Vt, 32, 72, 32);
      o2 = WMMA_BF16(pa0, bv0, o2); o2 = WMMA_BF16(pa1, bv1, o2);
    }
    {
      v16bf bv0 = frag_b(Vt, 48, 72, 0), bv1 = frag_b(Vt, 48, 72, 32);
      o3 = WMMA_BF16(pa0, bv0, o3); o3 = WMMA_BF16(pa1, bv1, o3);
    }
    __syncthreads();
  }

  // ---- epilogue: single row-sum reduction, O *= 1/l, store bf16 ----
  float rinv[8];
#pragma unroll
  for (int j = 0; j < 8; ++j) rinv[j] = 1.0f / rowsum16(lpart[j]);
#pragma unroll
  for (int ct = 0; ct < 4; ++ct) {
    const v8f& oc = ct == 0 ? o0 : ct == 1 ? o1 : ct == 2 ? o2 : o3;
    const int col = h * HEAD_DIM + ct * 16 + (l & 15);
#pragma unroll
    for (int j = 0; j < 8; ++j) {
      const int qr = q0 + j + half * 8;
      O[((size_t)b * NQ + qr) * DIM + col] = (__bf16)(oc[j] * rinv[j]);
    }
  }
}

// ---------------------------------------------------------------------------
// Launch: 3 projection GEMMs -> flash attention -> output GEMM
// ---------------------------------------------------------------------------
extern "C" void kernel_launch(void* const* d_in, const int* in_sizes, int n_in,
                              void* d_out, int out_size, void* d_ws, size_t ws_size,
                              hipStream_t stream) {
  const float* query    = (const float*)d_in[0];
  const float* key      = (const float*)d_in[1];
  const float* value    = (const float*)d_in[2];
  const float* att_bias = (const float*)d_in[3];
  const float* Wq = (const float*)d_in[4];
  const float* bq = (const float*)d_in[5];
  const float* Wk = (const float*)d_in[6];
  const float* bk = (const float*)d_in[7];
  const float* Wv = (const float*)d_in[8];
  const float* bv = (const float*)d_in[9];
  const float* Wo = (const float*)d_in[10];
  const float* bo = (const float*)d_in[11];
  float* out = (float*)d_out;

  const size_t M = 2 * NQ;                 // 4096 flattened rows
  const size_t bufElems = M * DIM;         // 4M bf16 elems = 8 MB each
  __bf16* qws = (__bf16*)d_ws;
  __bf16* kws = qws + bufElems;
  __bf16* vws = kws + bufElems;
  __bf16* aws = vws + bufElems;

  dim3 tb(256);
  dim3 gg((unsigned)(M / 128), DIM / 64, 1);    // (32, 16)

  gemm_bias_kernel<false, true><<<gg, tb, 0, stream>>>(query, Wq, bq, qws, (int)M, DIM, DIM);
  gemm_bias_kernel<false, true><<<gg, tb, 0, stream>>>(key,   Wk, bk, kws, (int)M, DIM, DIM);
  gemm_bias_kernel<false, true><<<gg, tb, 0, stream>>>(value, Wv, bv, vws, (int)M, DIM, DIM);

  dim3 ga(NQ / 128, HEADS, 2);                  // (16, 16, 2)
  attn_kernel<<<ga, tb, 0, stream>>>(qws, kws, vws, att_bias, aws);

  gemm_bias_kernel<true, false><<<gg, tb, 0, stream>>>(aws, Wo, bo, out, (int)M, DIM, DIM);
}
